// GATConv_39599598469258
// MI455X (gfx1250) — matrix operand order repaired and checked by reading, not compiled, not run on hardware
//
#include <hip/hip_runtime.h>
#include <hip/hip_bf16.h>
#include <cstdint>

typedef _Float16 v16h __attribute__((ext_vector_type(16)));
typedef _Float16 v8h  __attribute__((ext_vector_type(8)));
typedef float    v8f  __attribute__((ext_vector_type(8)));
typedef float    v4f  __attribute__((ext_vector_type(4)));

#define D_IN 128
#define HC   128
#define NH   4
#define CH   32
#define NEG_SLOPE 0.2f
#define SM_EPS 1e-16f
#define TILE_M 16
#define KPAD 136   // padded f16 row length (272B) -> per-lane b128 loads hit distinct banks

// ---------------- order-preserving float<->uint for atomic segment-max ----------------
__device__ __forceinline__ unsigned int fkey(float f) {
  unsigned int b = __float_as_uint(f);
  return (b & 0x80000000u) ? ~b : (b | 0x80000000u);
}
__device__ __forceinline__ float funkey(unsigned int k) {
  return __uint_as_float((k & 0x80000000u) ? (k ^ 0x80000000u) : ~k);
}

// ---------------- kernel 1: h = x @ W via v_wmma_f32_16x16x32_f16 ----------------
__global__ __launch_bounds__(256) void gat_proj(const float* __restrict__ x,
                                                const float* __restrict__ W,
                                                float* __restrict__ hout,
                                                int N, int num_tiles) {
  __shared__ __align__(16) _Float16 wlds[HC][KPAD];    // wlds[n][k] = W[k][n] (transposed)
  __shared__ __align__(16) _Float16 xlds[TILE_M][KPAD];

  const int t = threadIdx.x;
  // stage W (128x128 f32 -> f16, transposed) once per persistent block; b128 reads
  for (int i = t; i < D_IN * HC / 4; i += 256) {
    int k = i >> 5, c4 = (i & 31) * 4;
    v4f wv4 = *reinterpret_cast<const v4f*>(&W[(size_t)k * HC + c4]);
    wlds[c4 + 0][k] = (_Float16)wv4.x;
    wlds[c4 + 1][k] = (_Float16)wv4.y;
    wlds[c4 + 2][k] = (_Float16)wv4.z;
    wlds[c4 + 3][k] = (_Float16)wv4.w;
  }
  __syncthreads();

  const int lane = t & 31;
  const int wv   = t >> 5;       // wave id 0..7 -> 16-column slab
  const int lr   = lane & 15;
  const int half = lane >> 4;    // lane half selects K/M sub-block per ISA layout
  const int n    = wv * 16 + lr; // output column this lane owns in B/C/D frags

  for (int tile = blockIdx.x; tile < num_tiles; tile += gridDim.x) {
    const int r0 = tile * TILE_M;
    const bool full = (r0 + TILE_M) <= N;
    // stage x tile (16x128 f32 -> f16); x is streamed once -> non-temporal b128 loads
    if (full) {
      for (int i = t; i < TILE_M * D_IN / 4; i += 256) {
        int r = i >> 5, c4 = (i & 31) * 4;
        v4f xv = __builtin_nontemporal_load(
            reinterpret_cast<const v4f*>(&x[(size_t)(r0 + r) * D_IN + c4]));
        xlds[r][c4 + 0] = (_Float16)xv.x;
        xlds[r][c4 + 1] = (_Float16)xv.y;
        xlds[r][c4 + 2] = (_Float16)xv.z;
        xlds[r][c4 + 3] = (_Float16)xv.w;
      }
    } else {
      for (int i = t; i < TILE_M * D_IN; i += 256) {
        int r = i >> 7, c = i & 127;
        int row = r0 + r;
        xlds[r][c] = (_Float16)(row < N ? x[(size_t)row * D_IN + c] : 0.0f);
      }
    }
    __syncthreads();

    v8f acc = {};
#pragma unroll
    for (int k0 = 0; k0 < D_IN; k0 += 32) {
      union { v16h v; v8h p[2]; } a, b;
      // A 16x32 f16: lanes 0-15 rows, v0..3 = K(0..7 | 8..15 by half), v4..7 = +16
      a.p[0] = *reinterpret_cast<const v8h*>(&xlds[lr][k0 + half * 8]);
      a.p[1] = *reinterpret_cast<const v8h*>(&xlds[lr][k0 + 16 + half * 8]);
      // B 32x16 f16: lane = column n, K pairs packed; v0..3 = K(0..7 | 8..15 by half), v4..7 = +16
      b.p[0] = *reinterpret_cast<const v8h*>(&wlds[n][k0 + half * 8]);
      b.p[1] = *reinterpret_cast<const v8h*>(&wlds[n][k0 + 16 + half * 8]);
      acc = __builtin_amdgcn_wmma_f32_16x16x32_f16(false, a.v, false, b.v,
                                                   (short)0, acc, false, false);
    }
    // C/D layout: VGPR v -> row v (lanes 0-15) / v+8 (lanes 16-31), col = lane%16
    const int row0 = r0 + half * 8;
    if (full) {
#pragma unroll
      for (int v = 0; v < 8; ++v)
        hout[(size_t)(row0 + v) * HC + n] = acc[v];
    } else {
#pragma unroll
      for (int v = 0; v < 8; ++v) {
        int row = row0 + v;
        if (row < N) hout[(size_t)row * HC + n] = acc[v];
      }
    }
    __syncthreads();
  }
}

// ---------------- kernel 2: per-node logits + init (wave per node) ----------------
__global__ __launch_bounds__(256) void gat_logits(const float* __restrict__ h,
                                                  const float* __restrict__ att_src,
                                                  const float* __restrict__ att_dst,
                                                  const float* __restrict__ bias,
                                                  float* __restrict__ a_src,
                                                  float* __restrict__ a_dst,
                                                  unsigned int* __restrict__ mU,
                                                  float* __restrict__ sS,
                                                  float* __restrict__ out, int N) {
  const int wv = threadIdx.x >> 5, lane = threadIdx.x & 31;
  const int node = blockIdx.x * 8 + wv;
  if (node >= N) return;
  float ps[NH], pd[NH];
#pragma unroll
  for (int j = 0; j < NH; ++j) {
    float hv = h[(size_t)node * HC + j * CH + lane];
    ps[j] = hv * att_src[j * CH + lane];
    pd[j] = hv * att_dst[j * CH + lane];
    out[(size_t)node * HC + j * CH + lane] = bias[j * CH + lane];  // out starts at bias
  }
#pragma unroll
  for (int off = 16; off > 0; off >>= 1) {
#pragma unroll
    for (int j = 0; j < NH; ++j) {
      ps[j] += __shfl_down(ps[j], off, 32);
      pd[j] += __shfl_down(pd[j], off, 32);
    }
  }
  if (lane == 0) {
#pragma unroll
    for (int j = 0; j < NH; ++j) {
      a_src[(size_t)node * NH + j] = ps[j];
      a_dst[(size_t)node * NH + j] = pd[j];
      mU[(size_t)node * NH + j] = 0u;     // key 0 == ordering floor
      sS[(size_t)node * NH + j] = 0.0f;
    }
  }
}

// ---------------- kernel 3: edge logits + segment max (thread per edge) ----------------
__global__ __launch_bounds__(256) void gat_edge_max(const long long* __restrict__ srcI,
                                                    const long long* __restrict__ dstI,
                                                    int E, int Etot,
                                                    const float* __restrict__ a_src,
                                                    const float* __restrict__ a_dst,
                                                    float* __restrict__ elog,
                                                    unsigned int* __restrict__ mU) {
  const int i = blockIdx.x * 256 + threadIdx.x;
  if (i >= Etot) return;
  int s, d;
  if (i < E) { s = (int)srcI[i]; d = (int)dstI[i]; } else { s = d = i - E; }
  float4 as = *reinterpret_cast<const float4*>(&a_src[(size_t)s * NH]);
  float4 ad = *reinterpret_cast<const float4*>(&a_dst[(size_t)d * NH]);
  float ev[NH] = {as.x + ad.x, as.y + ad.y, as.z + ad.z, as.w + ad.w};
  float4 st;
#pragma unroll
  for (int j = 0; j < NH; ++j) {
    float e = ev[j];
    e = fmaxf(e, 0.0f) + NEG_SLOPE * fminf(e, 0.0f);  // LeakyReLU(0.2)
    ev[j] = e;
    atomicMax(&mU[(size_t)d * NH + j], fkey(e));
  }
  st.x = ev[0]; st.y = ev[1]; st.z = ev[2]; st.w = ev[3];
  *reinterpret_cast<float4*>(&elog[(size_t)i * NH]) = st;
}

// ---------------- kernel 4: exp + segment sum (thread per edge) ----------------
__global__ __launch_bounds__(256) void gat_edge_sum(const long long* __restrict__ dstI,
                                                    int E, int Etot,
                                                    float* __restrict__ elog,
                                                    const unsigned int* __restrict__ mU,
                                                    float* __restrict__ sS) {
  const int i = blockIdx.x * 256 + threadIdx.x;
  if (i >= Etot) return;
  int d = (i < E) ? (int)dstI[i] : (i - E);
  float4 e4 = *reinterpret_cast<const float4*>(&elog[(size_t)i * NH]);
  uint4  m4 = *reinterpret_cast<const uint4*>(&mU[(size_t)d * NH]);
  float ex[NH];
  ex[0] = __expf(e4.x - funkey(m4.x));
  ex[1] = __expf(e4.y - funkey(m4.y));
  ex[2] = __expf(e4.z - funkey(m4.z));
  ex[3] = __expf(e4.w - funkey(m4.w));
#pragma unroll
  for (int j = 0; j < NH; ++j) atomicAdd(&sS[(size_t)d * NH + j], ex[j]);
  float4 st = {ex[0], ex[1], ex[2], ex[3]};
  *reinterpret_cast<float4*>(&elog[(size_t)i * NH]) = st;  // overwrite with exp()
}

// ---------------- kernel 5: weighted aggregation (wave per edge) ----------------
__global__ __launch_bounds__(256) void gat_aggr(const long long* __restrict__ srcI,
                                                const long long* __restrict__ dstI,
                                                int E, int Etot,
                                                const float* __restrict__ h,
                                                const float* __restrict__ elog,
                                                const float* __restrict__ sS,
                                                float* __restrict__ out) {
  const int wv = threadIdx.x >> 5, lane = threadIdx.x & 31;
  const int i = blockIdx.x * 8 + wv;
  if (i >= Etot) return;
  int s, d;
  if (i < E) { s = (int)srcI[i]; d = (int)dstI[i]; } else { s = d = i - E; }
  float4 ex = *reinterpret_cast<const float4*>(&elog[(size_t)i * NH]);
  float4 sv = *reinterpret_cast<const float4*>(&sS[(size_t)d * NH]);
  float alpha[NH] = {ex.x / (sv.x + SM_EPS), ex.y / (sv.y + SM_EPS),
                     ex.z / (sv.z + SM_EPS), ex.w / (sv.w + SM_EPS)};
#pragma unroll
  for (int j = 0; j < NH; ++j) {
    float val = h[(size_t)s * HC + j * CH + lane] * alpha[j];
    atomicAdd(&out[(size_t)d * HC + j * CH + lane], val);
  }
}

extern "C" void kernel_launch(void* const* d_in, const int* in_sizes, int n_in,
                              void* d_out, int out_size, void* d_ws, size_t ws_size,
                              hipStream_t stream) {
  const float*     x       = (const float*)d_in[0];
  const long long* ei      = (const long long*)d_in[1];  // int64 per reference
  const float*     W       = (const float*)d_in[2];
  const float*     att_src = (const float*)d_in[3];
  const float*     att_dst = (const float*)d_in[4];
  const float*     bias    = (const float*)d_in[5];
  float*           out     = (float*)d_out;

  const int N    = in_sizes[0] / D_IN;
  const int E    = in_sizes[1] / 2;
  const int Etot = E + N;

  // workspace layout (floats)
  float* ws     = (float*)d_ws;
  float* h      = ws;                                   // N*128
  float* a_src  = h + (size_t)N * HC;                   // N*4
  float* a_dst  = a_src + (size_t)N * NH;               // N*4
  unsigned int* mU = (unsigned int*)(a_dst + (size_t)N * NH);  // N*4 (ordered keys)
  float* sS     = (float*)(mU + (size_t)N * NH);        // N*4
  float* elog   = sS + (size_t)N * NH;                  // Etot*4

  const long long* srcI = ei;
  const long long* dstI = ei + E;

  const int tiles = (N + TILE_M - 1) / TILE_M;
  const int nb = tiles < 512 ? tiles : 512;
  gat_proj<<<nb, 256, 0, stream>>>(x, W, h, N, tiles);
  gat_logits<<<(N + 7) / 8, 256, 0, stream>>>(h, att_src, att_dst, bias,
                                              a_src, a_dst, mU, sS, out, N);
  gat_edge_max<<<(Etot + 255) / 256, 256, 0, stream>>>(srcI, dstI, E, Etot,
                                                       a_src, a_dst, elog, mU);
  gat_edge_sum<<<(Etot + 255) / 256, 256, 0, stream>>>(dstI, E, Etot, elog, mU, sS);
  gat_aggr<<<(Etot + 7) / 8, 256, 0, stream>>>(srcI, dstI, E, Etot, h, elog, sS, out);
}
